// Pointnet2BackboneRefine_10075993276694
// MI455X (gfx1250) — matrix operand order repaired and checked by reading, not compiled, not run on hardware
//
#include <hip/hip_runtime.h>
#include <hip/hip_bf16.h>

typedef _Float16 v8h  __attribute__((ext_vector_type(8)));
typedef _Float16 v16h __attribute__((ext_vector_type(16)));
typedef float    v8f  __attribute__((ext_vector_type(8)));

#define NOBJ   4864
#define NSUP1  13056
#define M1 6912
#define M2 5888
#define M3 5376
#define M4 5120

// ---------------- WMMA fragment helpers (contiguous k order, A and B alike) --------

__device__ __forceinline__ v16h load_a_frag(const _Float16* __restrict__ p) {
  // A 16x32 f16: lane m=L%16, hi=L/16; elems 0..15 -> K=hi*16+e
  v8h a0 = *(const v8h*)(p);
  v8h a1 = *(const v8h*)(p + 8);
  v16h af;
#pragma unroll
  for (int e = 0; e < 8; e++) { af[e] = a0[e]; af[8 + e] = a1[e]; }
  return af;
}

// One GEMM layer: D(Cout x NT16*16) = relu(W(Cout x Kpad) * X + scale/bias), X/D in LDS
// LDS activation layout: lds[col*PITCH + k] (f16), so B-frag = 16 contiguous halves,
// D store = 8 contiguous halves per lane.
template<int PITCH, int NT16>
__device__ __forceinline__ void gemm_lds_layer(const _Float16* __restrict__ W,
    const float* __restrict__ S, const float* __restrict__ Bv,
    int Cout, int Kpad, int in_off, int out_off, _Float16* lds, int t)
{
  const int lane = t & 31, wave = t >> 5;
  const int ln = lane & 15, hi = lane >> 4;
  const int ntiles = (Cout >> 4) * NT16;
  for (int tt = wave; tt < ntiles; tt += 8) {
    const int mt = tt / NT16;
    const int nt = tt - mt * NT16;
    v8f acc;
#pragma unroll
    for (int e = 0; e < 8; e++) acc[e] = 0.f;
    const _Float16* wrow = W + (size_t)(mt * 16 + ln) * Kpad + hi * 16;
    const _Float16* bptr = lds + (size_t)(nt * 16 + ln) * PITCH + in_off + hi * 16;
    for (int kt = 0; kt < Kpad; kt += 32) {
      v16h af = load_a_frag(wrow + kt);
      v16h bf = *(const v16h*)(bptr + kt);
      acc = __builtin_amdgcn_wmma_f32_16x16x32_f16(false, af, false, bf,
                                                   (short)0, acc, false, false);
    }
    const int mr = mt * 16 + hi * 8;
    v8h r;
#pragma unroll
    for (int e = 0; e < 8; e++) {
      float v = acc[e] * S[mr + e] + Bv[mr + e];
      r[e] = (_Float16)(v > 0.f ? v : 0.f);
    }
    *(v8h*)(lds + (size_t)(nt * 16 + ln) * PITCH + out_off + mr) = r;
  }
}

// ---------------- stage 0: build support xyz + 20-channel features ------------

__global__ void prep_kernel(const float* __restrict__ pc, const float* __restrict__ cen,
    const float* __restrict__ cue, const float* __restrict__ matching,
    const int* __restrict__ msem, const float* __restrict__ floorh,
    const float* __restrict__ mcen, float* __restrict__ sup, float* __restrict__ feat)
{
  const int bb = blockIdx.y;
  const int n = blockIdx.x * blockDim.x + threadIdx.x;
  if (n >= NSUP1) return;
  const float* pcb  = pc  + (size_t)bb * 8192 * 4;
  const float* cenb = cen + (size_t)bb * 256 * 3;
  const float* cueb = cue + (size_t)bb * 4608 * 3;
  float* supb = sup + (size_t)bb * NSUP1 * 3;
  float* fb   = feat + (size_t)bb * 20 * NSUP1;
  const float fh = floorh[bb];
  float x, y, z, c0, c1; int sem;
  if (n < 256) {
    x = cenb[n*3]; y = cenb[n*3+1]; z = cenb[n*3+2];
    c0 = z - fh; c1 = mcen[bb*256 + n]; sem = msem[bb*NOBJ + n];
  } else if (n < NOBJ) {
    const int j = n - 256;
    x = cueb[j*3]; y = cueb[j*3+1]; z = cueb[j*3+2];
    c0 = z - fh; c1 = matching[bb*4608 + j]; sem = msem[bb*NOBJ + n];
  } else {
    const int p = n - NOBJ;
    x = pcb[p*4]; y = pcb[p*4+1]; z = pcb[p*4+2];
    c0 = pcb[p*4+3]; c1 = 0.f; sem = -1;
  }
  supb[n*3] = x; supb[n*3+1] = y; supb[n*3+2] = z;
  fb[0*NSUP1 + n] = c0;
  fb[1*NSUP1 + n] = c1;
#pragma unroll
  for (int k = 0; k < 18; k++) fb[(2 + k)*NSUP1 + n] = (sem == k) ? 1.f : 0.f;
}

// ---------------- weight fp32 -> f16 (K zero-padded to mult of 32) ------------

__global__ void convw_kernel(const float* __restrict__ w, _Float16* __restrict__ dst,
                             int Cin, int Kpad, int total)
{
  const int i = blockIdx.x * blockDim.x + threadIdx.x;
  if (i >= total) return;
  const int m = i / Kpad, k = i - m * Kpad;
  dst[i] = (k < Cin) ? (_Float16)w[m * Cin + k] : (_Float16)0.f;
}

// ---------------- farthest point sampling (1 WG per batch) -------------------

__global__ __launch_bounds__(1024) void fps_kernel(const float* __restrict__ sup1,
                                                   int* __restrict__ inds)
{
  __shared__ float mind[8192];
  __shared__ float rv[1024];
  __shared__ int   ri[1024];
  const int bb = blockIdx.x;
  const float* xyz = sup1 + (size_t)bb * NSUP1 * 3 + NOBJ * 3;
  int* indb = inds + bb * 2048;
  const int t = threadIdx.x;
  for (int i = t; i < 8192; i += 1024) mind[i] = 1e10f;
  __syncthreads();
  int far = 0;
  for (int s = 0; s < 2048; s++) {
    if (t == 0) indb[s] = far;
    const float cx = xyz[far*3], cy = xyz[far*3+1], cz = xyz[far*3+2];
    float bv = -1.f; int bi = 0;
    for (int i = t; i < 8192; i += 1024) {
      const float dx = xyz[i*3]-cx, dy = xyz[i*3+1]-cy, dz = xyz[i*3+2]-cz;
      const float d = dx*dx + dy*dy + dz*dz;
      const float m = fminf(mind[i], d);
      mind[i] = m;
      if (m > bv) { bv = m; bi = i; }
    }
    rv[t] = bv; ri[t] = bi;
    __syncthreads();
    for (int off = 512; off > 0; off >>= 1) {
      if (t < off) {
        const float v2 = rv[t+off]; const int i2 = ri[t+off];
        if (v2 > rv[t] || (v2 == rv[t] && i2 < ri[t])) { rv[t] = v2; ri[t] = i2; }
      }
      __syncthreads();
    }
    far = ri[0];
    __syncthreads();
  }
}

// ---------------- build new_xyz for all 4 SA stages (index chaining) ----------

__global__ void newxyz_kernel(const float* __restrict__ sup1,
    const int* __restrict__ i1, const int* __restrict__ i2,
    const int* __restrict__ i3, const int* __restrict__ i4,
    float* __restrict__ x1, float* __restrict__ x2,
    float* __restrict__ x3, float* __restrict__ x4)
{
  const int bb = blockIdx.y;
  const int r = blockIdx.x * blockDim.x + threadIdx.x;
  const float* supb = sup1 + (size_t)bb * NSUP1 * 3;
  const float* xyzb = supb + NOBJ * 3;
  const int* a1 = i1 + bb * 2048;
  const int* a2 = i2 + bb * 1024;
  const int* a3 = i3 + bb * 512;
  const int* a4 = i4 + bb * 256;
  float* dst; const float* src;
  if (r < M1) {
    dst = x1 + (size_t)bb*M1*3 + (size_t)r*3;
    src = (r < NOBJ) ? supb + (size_t)r*3 : xyzb + (size_t)a1[r-NOBJ]*3;
  } else if (r < M1+M2) {
    const int q = r - M1;
    dst = x2 + (size_t)bb*M2*3 + (size_t)q*3;
    src = (q < NOBJ) ? supb + (size_t)q*3 : xyzb + (size_t)a1[a2[q-NOBJ]]*3;
  } else if (r < M1+M2+M3) {
    const int q = r - M1 - M2;
    dst = x3 + (size_t)bb*M3*3 + (size_t)q*3;
    src = (q < NOBJ) ? supb + (size_t)q*3 : xyzb + (size_t)a1[a2[a3[q-NOBJ]]]*3;
  } else if (r < M1+M2+M3+M4) {
    const int q = r - M1 - M2 - M3;
    dst = x4 + (size_t)bb*M4*3 + (size_t)q*3;
    src = (q < NOBJ) ? supb + (size_t)q*3 : xyzb + (size_t)a1[a2[a3[a4[q-NOBJ]]]]*3;
  } else return;
  dst[0] = src[0]; dst[1] = src[1]; dst[2] = src[2];
}

// ---------------- ball query: first nsample in-radius indices (ascending) -----

__global__ void ballq_kernel(const float* __restrict__ sup, int Nsup, size_t supStride,
    const float* __restrict__ cen, int M, size_t cenStride,
    float r2, int nsample, int* __restrict__ idx, size_t idxStride)
{
  const int bb = blockIdx.y;
  const int m = blockIdx.x * blockDim.x + threadIdx.x;
  if (m >= M) return;
  const float* sb = sup + (size_t)bb * supStride;
  const float* cb = cen + (size_t)bb * cenStride;
  int* ib = idx + (size_t)bb * idxStride + (size_t)m * nsample;
  const float cx = cb[m*3], cy = cb[m*3+1], cz = cb[m*3+2];
  int cnt = 0, first = -1;
  for (int i = 0; i < Nsup; i++) {
    const float dx = sb[i*3]-cx, dy = sb[i*3+1]-cy, dz = sb[i*3+2]-cz;
    const float d2 = dx*dx + dy*dy + dz*dz;
    if (d2 < r2) {
      if (first < 0) first = i;
      ib[cnt++] = i;
      if (cnt == nsample) break;
    }
  }
  const int pv = (first < 0) ? 0 : first;
  for (int j = cnt; j < nsample; j++) ib[j] = pv;
}

// ---------------- fused SA: gather -> 3x (WMMA GEMM+relu) -> maxpool ----------

#define SA_NT 64
#define SA_PITCH 432   // halves per column; arena rows (mult of 16)

__global__ __launch_bounds__(256) void sa_mlp_kernel(
    const float* __restrict__ sup, size_t supStride,
    const float* __restrict__ cen, size_t cenStride,
    const float* __restrict__ feats, int Cfeat, int featRow, size_t featStride,
    const int* __restrict__ idx, size_t idxStride,
    int nshift, float inv_radius, int Kpad0,
    const _Float16* __restrict__ w0, const float* __restrict__ s0p, const float* __restrict__ b0p, int C1,
    const _Float16* __restrict__ w1, const float* __restrict__ s1p, const float* __restrict__ b1p, int C2,
    const _Float16* __restrict__ w2, const float* __restrict__ s2p, const float* __restrict__ b2p, int C3,
    float* __restrict__ out, size_t outStride, int M)
{
  __shared__ __attribute__((aligned(32))) _Float16 lds[SA_NT * SA_PITCH];
  const int bb = blockIdx.y;
  const int t = threadIdx.x;
  const int colBase = blockIdx.x * SA_NT;
  const float* supb = sup + (size_t)bb * supStride;
  const float* cenb = cen + (size_t)bb * cenStride;
  const float* fbb  = feats + (size_t)bb * featStride;
  const int* idxb   = idx + (size_t)bb * idxStride;

  // non-overlapping ping-pong offsets (rows): verified per-stage <= SA_PITCH
  const int offA = Kpad0;
  const int offB = (C2 <= Kpad0) ? 0 : (Kpad0 + C1);
  const int offC = offB + C2;

  // gather x0 = [(p-c)/r ; feats[idx]] -> lds[col][k] (f16, zero-padded K)
  {
    const int col  = t & 63;
    const int gcol = colBase + col;
    const int m  = gcol >> nshift;
    const int iv = idxb[gcol];
    const float g0 = (supb[iv*3+0] - cenb[m*3+0]) * inv_radius;
    const float g1 = (supb[iv*3+1] - cenb[m*3+1]) * inv_radius;
    const float g2 = (supb[iv*3+2] - cenb[m*3+2]) * inv_radius;
    const int Cin = Cfeat + 3;
    for (int k = t >> 6; k < Kpad0; k += 4) {
      float v;
      if (k == 0) v = g0;
      else if (k == 1) v = g1;
      else if (k == 2) v = g2;
      else if (k < Cin) v = fbb[(size_t)(k - 3) * featRow + iv];
      else v = 0.f;
      lds[(size_t)col * SA_PITCH + k] = (_Float16)v;
    }
  }
  __syncthreads();
  gemm_lds_layer<SA_PITCH, 4>(w0, s0p, b0p, C1, Kpad0, 0,    offA, lds, t);
  __syncthreads();
  gemm_lds_layer<SA_PITCH, 4>(w1, s1p, b1p, C2, C1,    offA, offB, lds, t);
  __syncthreads();
  gemm_lds_layer<SA_PITCH, 4>(w2, s2p, b2p, C3, C2,    offB, offC, lds, t);
  __syncthreads();

  // maxpool over nsample columns per center -> out[o*M + m] (f32)
  const int nsample = 1 << nshift;
  const int cpt = SA_NT >> nshift;
  const int total = cpt * C3;
  float* ob = out + (size_t)bb * outStride;
  for (int i = t; i < total; i += 256) {
    const int c = i / C3;
    const int o = i - c * C3;
    const _Float16* p = lds + (size_t)(c * nsample) * SA_PITCH + offC + o;
    float mv = (float)p[0];
    for (int j = 1; j < nsample; j++) {
      const float v = (float)p[(size_t)j * SA_PITCH];
      mv = v > mv ? v : mv;
    }
    ob[(size_t)o * M + ((colBase >> nshift) + c)] = mv;
  }
}

// ---------------- FP: 3-NN (stable, squared-dist weights) ---------------------

__global__ void knn3_kernel(const float* __restrict__ q, size_t qStride, int N1,
    const float* __restrict__ ref, size_t refStride, int N2,
    int* __restrict__ oi, float* __restrict__ ow, size_t oStride)
{
  const int bb = blockIdx.y;
  const int p = blockIdx.x * blockDim.x + threadIdx.x;
  if (p >= N1) return;
  const float* qb = q + (size_t)bb * qStride;
  const float* rb = ref + (size_t)bb * refStride;
  const float qx = qb[p*3], qy = qb[p*3+1], qz = qb[p*3+2];
  float d0 = 3.4e38f, d1 = 3.4e38f, d2v = 3.4e38f;
  int i0 = 0, i1 = 0, i2 = 0;
  for (int i = 0; i < N2; i++) {
    const float dx = rb[i*3]-qx, dy = rb[i*3+1]-qy, dz = rb[i*3+2]-qz;
    const float d = dx*dx + dy*dy + dz*dz;
    if (d < d0)      { d2v = d1; i2 = i1; d1 = d0; i1 = i0; d0 = d; i0 = i; }
    else if (d < d1) { d2v = d1; i2 = i1; d1 = d;  i1 = i; }
    else if (d < d2v){ d2v = d;  i2 = i; }
  }
  float w0 = 1.f/(d0 + 1e-8f), w1 = 1.f/(d1 + 1e-8f), w2 = 1.f/(d2v + 1e-8f);
  const float s = w0 + w1 + w2;
  w0 /= s; w1 /= s; w2 /= s;
  int* ob = oi + (size_t)bb * oStride + (size_t)p * 3;
  float* wb = ow + (size_t)bb * oStride + (size_t)p * 3;
  ob[0] = i0; ob[1] = i1; ob[2] = i2;
  wb[0] = w0; wb[1] = w1; wb[2] = w2;
}

__global__ void interp_kernel(const float* __restrict__ fsrc, int fRow, size_t fStride,
    const int* __restrict__ ki, const float* __restrict__ kw, size_t kStride,
    float* __restrict__ dst, int N1, size_t dStride)
{
  const int bb = blockIdx.y;
  const int ch = blockIdx.z;
  const int p = blockIdx.x * blockDim.x + threadIdx.x;
  if (p >= N1) return;
  const int* ib = ki + (size_t)bb * kStride + (size_t)p * 3;
  const float* wb = kw + (size_t)bb * kStride + (size_t)p * 3;
  const float* fb = fsrc + (size_t)bb * fStride + (size_t)ch * fRow;
  dst[(size_t)bb * dStride + (size_t)ch * N1 + p] =
      fb[ib[0]] * wb[0] + fb[ib[1]] * wb[1] + fb[ib[2]] * wb[2];
}

// ---------------- fused FP MLP: gather(512ch) -> 2x WMMA GEMM+relu ------------

#define FP_NT 32
#define FP_PITCH 768

__global__ __launch_bounds__(256) void fp_mlp_kernel(
    const float* __restrict__ interp, size_t interpStride, int N,
    const float* __restrict__ feat1, int f1Row, size_t f1Stride,
    const _Float16* __restrict__ w0, const float* __restrict__ s0p, const float* __restrict__ b0p,
    const _Float16* __restrict__ w1, const float* __restrict__ s1p, const float* __restrict__ b1p,
    float* __restrict__ out, int outRow, size_t outStride)
{
  __shared__ __attribute__((aligned(32))) _Float16 lds[FP_NT * FP_PITCH];
  const int bb = blockIdx.y, t = threadIdx.x;
  const int colBase = blockIdx.x * FP_NT;
  const float* ib = interp + (size_t)bb * interpStride;
  const float* fb = feat1 + (size_t)bb * f1Stride;
  {
    const int col = t & 31;
    const int p = colBase + col;
    for (int k = t >> 5; k < 512; k += 8) {
      const float v = (k < 256) ? ib[(size_t)k * N + p]
                                : fb[(size_t)(k - 256) * f1Row + p];
      lds[(size_t)col * FP_PITCH + k] = (_Float16)v;
    }
  }
  __syncthreads();
  gemm_lds_layer<FP_PITCH, 2>(w0, s0p, b0p, 256, 512, 0, 512, lds, t);
  __syncthreads();
  // last layer: WMMA then direct f32 store to global
  {
    const int lane = t & 31, wave = t >> 5, ln = lane & 15, hi = lane >> 4;
    float* ob = out + (size_t)bb * outStride;
    for (int tt = wave; tt < 32; tt += 8) {
      const int mt = tt >> 1, nt = tt & 1;
      v8f acc;
#pragma unroll
      for (int e = 0; e < 8; e++) acc[e] = 0.f;
      const _Float16* wrow = w1 + (size_t)(mt * 16 + ln) * 256 + hi * 16;
      const _Float16* bptr = lds + (size_t)(nt * 16 + ln) * FP_PITCH + 512 + hi * 16;
      for (int kt = 0; kt < 256; kt += 32) {
        v16h af = load_a_frag(wrow + kt);
        v16h bf = *(const v16h*)(bptr + kt);
        acc = __builtin_amdgcn_wmma_f32_16x16x32_f16(false, af, false, bf,
                                                     (short)0, acc, false, false);
      }
      const int mr = mt * 16 + hi * 8;
      const int p2 = colBase + nt * 16 + ln;
#pragma unroll
      for (int e = 0; e < 8; e++) {
        const float v = acc[e] * s1p[mr + e] + b1p[mr + e];
        ob[(size_t)(mr + e) * outRow + p2] = v > 0.f ? v : 0.f;
      }
    }
  }
}

// ---------------- tail: xyz2[:, :4864] and sa1_inds to d_out ------------------

__global__ void tail_kernel(const float* __restrict__ x2, const int* __restrict__ inds1,
                            float* __restrict__ out)
{
  const int bb = blockIdx.y;
  const int g = blockIdx.x * blockDim.x + threadIdx.x;
  const size_t OFF2 = (size_t)2 * 256 * 4864;
  const size_t OFF3 = OFF2 + (size_t)2 * 4864 * 3;
  if (g < 4864 * 3) {
    out[OFF2 + (size_t)bb * 4864 * 3 + g] = x2[(size_t)bb * M2 * 3 + g];
  } else {
    const int j = g - 4864 * 3;
    if (j < 2048) out[OFF3 + (size_t)bb * 2048 + j] = (float)inds1[bb * 2048 + j];
  }
}

// ------------------------------- host ----------------------------------------

extern "C" void kernel_launch(void* const* d_in, const int* in_sizes, int n_in,
                              void* d_out, int out_size, void* d_ws, size_t ws_size,
                              hipStream_t stream)
{
  (void)in_sizes; (void)n_in; (void)out_size; (void)ws_size;
  const float* pc       = (const float*)d_in[0];
  const float* cenp     = (const float*)d_in[1];
  const float* cuep     = (const float*)d_in[2];
  const float* matching = (const float*)d_in[3];
  const int*   msem     = (const int*)d_in[4];
  const float* floorh   = (const float*)d_in[5];
  const float* mcen     = (const float*)d_in[6];
  const int* sa2i = (const int*)d_in[7];
  const int* sa3i = (const int*)d_in[8];
  const int* sa4i = (const int*)d_in[9];

  char* ws = (char*)d_ws;
  size_t cur = 0;
  auto alloc = [&](size_t bytes) -> char* {
    char* p = ws + cur;
    cur = (cur + bytes + 255) & ~(size_t)255;
    return p;
  };

  _Float16* wq = (_Float16*)alloc((size_t)649216 * 2);
  float* sup1  = (float*)alloc((size_t)2 * 39168 * 4);
  float* feat0 = (float*)alloc((size_t)2 * 261120 * 4);
  int*   inds1 = (int*)  alloc((size_t)2 * 2048 * 4);
  float* x1    = (float*)alloc((size_t)2 * 20736 * 4);
  float* x2    = (float*)alloc((size_t)2 * 17664 * 4);
  float* x3    = (float*)alloc((size_t)2 * 16128 * 4);
  float* x4    = (float*)alloc((size_t)2 * 15360 * 4);
  int*   idx1  = (int*)  alloc((size_t)2 * 442368 * 4);
  int*   idx2  = (int*)  alloc((size_t)2 * 188416 * 4);
  int*   idx3  = (int*)  alloc((size_t)2 * 86016 * 4);
  int*   idx4  = (int*)  alloc((size_t)2 * 81920 * 4);
  float* f1    = (float*)alloc((size_t)2 * 884736 * 4);
  float* f2    = (float*)alloc((size_t)2 * 1507328 * 4);
  float* f3    = (float*)alloc((size_t)2 * 1376256 * 4);
  float* f4    = (float*)alloc((size_t)2 * 1310720 * 4);
  int*   kni   = (int*)  alloc((size_t)2 * 16128 * 4);
  float* knw   = (float*)alloc((size_t)2 * 16128 * 4);
  float* itp   = (float*)alloc((size_t)2 * 1376256 * 4);
  float* fp1f  = (float*)alloc((size_t)2 * 1376256 * 4);

  struct WD { int src; size_t off; int Cin; int Kpad; int Cout; };
  const WD wd[16] = {
    {10,      0,  23,  32,  64}, {13,   2048,  64,  64,  64}, {16,   6144,  64,  64, 128},
    {19,  14336, 131, 160, 128}, {22,  34816, 128, 128, 128}, {25,  51200, 128, 128, 256},
    {28,  83968, 259, 288, 128}, {31, 120832, 128, 128, 128}, {34, 137216, 128, 128, 256},
    {37, 169984, 259, 288, 128}, {40, 206848, 128, 128, 128}, {43, 223232, 128, 128, 256},
    {46, 256000, 512, 512, 256}, {49, 387072, 256, 256, 256},
    {52, 452608, 512, 512, 256}, {55, 583680, 256, 256, 256},
  };
  for (int i = 0; i < 16; i++) {
    const int total = wd[i].Cout * wd[i].Kpad;
    convw_kernel<<<dim3((total + 255) / 256), dim3(256), 0, stream>>>(
        (const float*)d_in[wd[i].src], wq + wd[i].off, wd[i].Cin, wd[i].Kpad, total);
  }

  prep_kernel<<<dim3(51, 2), 256, 0, stream>>>(pc, cenp, cuep, matching, msem,
                                               floorh, mcen, sup1, feat0);
  fps_kernel<<<2, 1024, 0, stream>>>(sup1, inds1);
  newxyz_kernel<<<dim3(91, 2), 256, 0, stream>>>(sup1, inds1, sa2i, sa3i, sa4i,
                                                 x1, x2, x3, x4);

  ballq_kernel<<<dim3(27, 2), 256, 0, stream>>>(sup1, 13056, 39168, x1, M1, 20736, 0.04f, 64, idx1, 442368);
  ballq_kernel<<<dim3(23, 2), 256, 0, stream>>>(x1, M1, 20736, x2, M2, 17664, 0.16f, 32, idx2, 188416);
  ballq_kernel<<<dim3(21, 2), 256, 0, stream>>>(x2, M2, 17664, x3, M3, 16128, 0.64f, 16, idx3, 86016);
  ballq_kernel<<<dim3(20, 2), 256, 0, stream>>>(x3, M3, 16128, x4, M4, 15360, 1.44f, 16, idx4, 81920);

  sa_mlp_kernel<<<dim3(6912, 2), 256, 0, stream>>>(
      sup1, 39168, x1, 20736, feat0, 20, 13056, 261120, idx1, 442368, 6, 5.0f, 32,
      wq + 0,    (const float*)d_in[11], (const float*)d_in[12], 64,
      wq + 2048, (const float*)d_in[14], (const float*)d_in[15], 64,
      wq + 6144, (const float*)d_in[17], (const float*)d_in[18], 128,
      f1, 884736, M1);
  sa_mlp_kernel<<<dim3(2944, 2), 256, 0, stream>>>(
      x1, 20736, x2, 17664, f1, 128, M1, 884736, idx2, 188416, 5, 2.5f, 160,
      wq + 14336, (const float*)d_in[20], (const float*)d_in[21], 128,
      wq + 34816, (const float*)d_in[23], (const float*)d_in[24], 128,
      wq + 51200, (const float*)d_in[26], (const float*)d_in[27], 256,
      f2, 1507328, M2);
  sa_mlp_kernel<<<dim3(1344, 2), 256, 0, stream>>>(
      x2, 17664, x3, 16128, f2, 256, M2, 1507328, idx3, 86016, 4, 1.25f, 288,
      wq + 83968,  (const float*)d_in[29], (const float*)d_in[30], 128,
      wq + 120832, (const float*)d_in[32], (const float*)d_in[33], 128,
      wq + 137216, (const float*)d_in[35], (const float*)d_in[36], 256,
      f3, 1376256, M3);
  sa_mlp_kernel<<<dim3(1280, 2), 256, 0, stream>>>(
      x3, 16128, x4, 15360, f3, 256, M3, 1376256, idx4, 81920, 4, 1.0f / 1.2f, 288,
      wq + 169984, (const float*)d_in[38], (const float*)d_in[39], 128,
      wq + 206848, (const float*)d_in[41], (const float*)d_in[42], 128,
      wq + 223232, (const float*)d_in[44], (const float*)d_in[45], 256,
      f4, 1310720, M4);

  // FP1: xyz3 (5376) <- xyz4[4864:] (256), feats f4[:, 4864:]
  knn3_kernel<<<dim3(21, 2), 256, 0, stream>>>(x3, 16128, M3, x4 + 4864 * 3, 15360, 256, kni, knw, 16128);
  interp_kernel<<<dim3(21, 2, 256), 256, 0, stream>>>(f4 + 4864, M4, 1310720, kni, knw, 16128, itp, M3, 1376256);
  fp_mlp_kernel<<<dim3(168, 2), 256, 0, stream>>>(
      itp, 1376256, M3, f3, M3, 1376256,
      wq + 256000, (const float*)d_in[47], (const float*)d_in[48],
      wq + 387072, (const float*)d_in[50], (const float*)d_in[51],
      fp1f, M3, 1376256);

  // FP2: xyz2[:4864] <- xyz3[4864:] (512), feats fp1_feat[:, 4864:]
  knn3_kernel<<<dim3(19, 2), 256, 0, stream>>>(x2, 17664, 4864, x3 + 4864 * 3, 16128, 512, kni, knw, 16128);
  interp_kernel<<<dim3(19, 2, 256), 256, 0, stream>>>(fp1f + 4864, M3, 1376256, kni, knw, 16128, itp, 4864, 1376256);
  float* outp = (float*)d_out;
  fp_mlp_kernel<<<dim3(152, 2), 256, 0, stream>>>(
      itp, 1376256, 4864, f2, M2, 1507328,
      wq + 452608, (const float*)d_in[53], (const float*)d_in[54],
      wq + 583680, (const float*)d_in[56], (const float*)d_in[57],
      outp, 4864, (size_t)256 * 4864);

  tail_kernel<<<dim3(65, 2), 256, 0, stream>>>(x2, inds1, outp);
}
